// SparseAutoencoder_57526791962759
// MI455X (gfx1250) — matrix operand order repaired and checked by reading, not compiled, not run on hardware
//
#include <hip/hip_runtime.h>
#include <cstddef>

#define D_IN   768
#define D_HID  16384
#define TOPK   32
#define ROWS   16          // rows (M) per block tile
#define CHUNK  128         // hidden cols per block iteration (8 waves * 16)
#define XS_STRIDE   (D_IN + 4)   // f32 path pad (772 % 64 != 0)
#define XSB_STRIDE  (D_IN + 8)   // bf16 path pad: 776*2B = 1552B = 388 dw, %64 = 4
#define ACT_STRIDE  (CHUNK + 4)

typedef __attribute__((ext_vector_type(2)))  float          v2f;
typedef __attribute__((ext_vector_type(8)))  float          v8f;
typedef __attribute__((ext_vector_type(16))) __bf16         v16bf;
typedef __attribute__((ext_vector_type(2)))  unsigned int   u32x2;
typedef __attribute__((ext_vector_type(4)))  unsigned int   u32x4;

struct Frag32 { u32x4 lo, hi; };   // 32B container for a v16bf fragment

__device__ __forceinline__ unsigned short bf16_rne(float f) {
  unsigned int u = __builtin_bit_cast(unsigned int, f);
  u += 0x7FFFu + ((u >> 16) & 1u);          // round to nearest even
  return (unsigned short)(u >> 16);
}
__device__ __forceinline__ float bf16_to_f32(unsigned short h) {
  return __builtin_bit_cast(float, (unsigned int)h << 16);
}

// ---------------------------------------------------------------------------
// Split W_enc (f32) into bf16 hi/lo planes (one-time, bandwidth bound)
// ---------------------------------------------------------------------------
__global__ __launch_bounds__(256) void wenc_split_kernel(
    const float* __restrict__ W, unsigned short* __restrict__ Whi,
    unsigned short* __restrict__ Wlo)
{
  const size_t i0 = ((size_t)blockIdx.x * 256 + threadIdx.x) * 4;
  float4 v = *(const float4*)(W + i0);
  unsigned short h[4], l[4];
  float vv[4] = {v.x, v.y, v.z, v.w};
  #pragma unroll
  for (int j = 0; j < 4; j++) {
    h[j] = bf16_rne(vv[j]);
    l[j] = bf16_rne(vv[j] - bf16_to_f32(h[j]));
  }
  u32x2 ph = { (unsigned)h[0] | ((unsigned)h[1] << 16),
               (unsigned)h[2] | ((unsigned)h[3] << 16) };
  u32x2 pl = { (unsigned)l[0] | ((unsigned)l[1] << 16),
               (unsigned)l[2] | ((unsigned)l[3] << 16) };
  *(u32x2*)(Whi + i0) = ph;
  *(u32x2*)(Wlo + i0) = pl;
}

// ---------------------------------------------------------------------------
// Tiled transpose: W_dec [768][16384] -> W_dec^T [16384][768] (in workspace)
// ---------------------------------------------------------------------------
__global__ __launch_bounds__(256) void wdec_transpose_kernel(
    const float* __restrict__ Wdec, float* __restrict__ WdT)
{
  __shared__ float tile[32][33];
  const int h0 = blockIdx.x * 32;
  const int i0 = blockIdx.y * 32;
  const int tx = threadIdx.x;   // 32
  const int ty = threadIdx.y;   // 8
  #pragma unroll
  for (int r = ty; r < 32; r += 8)
    tile[r][tx] = Wdec[(size_t)(i0 + r) * D_HID + (h0 + tx)];
  __syncthreads();
  #pragma unroll
  for (int r = ty; r < 32; r += 8)
    WdT[(size_t)(h0 + r) * D_IN + (i0 + tx)] = tile[tx][r];
}

// ---------------------------------------------------------------------------
// Fast path: split-bf16 (3-product) WMMA encode + parallel-filter top-k
//            + coalesced sparse decode
// ---------------------------------------------------------------------------
__global__ __launch_bounds__(256) void sae_fused_bf16_kernel(
    const float* __restrict__ x,
    const unsigned short* __restrict__ Whi,   // bf16 hi plane [H][768]
    const unsigned short* __restrict__ Wlo,   // bf16 lo plane [H][768]
    const float* __restrict__ b_enc,
    const float* __restrict__ WdT,            // W_dec^T [H][768]
    const float* __restrict__ b_dec,
    float* __restrict__ out)
{
  __shared__ unsigned short xs_hi[ROWS][XSB_STRIDE];  // ~24.8 KB each
  __shared__ unsigned short xs_lo[ROWS][XSB_STRIDE];
  __shared__ float act[ROWS][ACT_STRIDE];
  __shared__ float topv[ROWS][TOPK];
  __shared__ int   topi[ROWS][TOPK];
  __shared__ float rowmin[ROWS];                       // current 32nd-largest
  __shared__ unsigned char candb[ROWS][16];            // 128-bit candidate mask

  const int tid  = (int)threadIdx.x;
  const int wave = tid >> 5;
  const int lane = tid & 31;
  const int r0   = (int)blockIdx.x * ROWS;

  // ---- stage x tile, splitting to bf16 hi/lo ----
  {
    const float* src = x + (size_t)r0 * D_IN;
    for (int idx = tid; idx < ROWS * D_IN; idx += 256) {
      int r = idx / D_IN;
      int ccol = idx - r * D_IN;
      float v = src[idx];
      unsigned short h = bf16_rne(v);
      xs_hi[r][ccol] = h;
      xs_lo[r][ccol] = bf16_rne(v - bf16_to_f32(h));
    }
  }
  for (int idx = tid; idx < ROWS * TOPK; idx += 256) {
    ((float*)topv)[idx] = -1.0f;
    ((int*)topi)[idx]   = 0;
  }
  if (tid < ROWS) rowmin[tid] = -1.0f;
  __syncthreads();

  // 16-bit WMMA fragment geometry (ISA §7.12.2)
  const int nn   = lane & 15;        // A row, B column, C column
  const int half = lane >> 4;
  const int mbase = half << 3;       // C rows: 0..7 or 8..15
  const int aK0  = 8 * half;         // A: K runs [8h, 8h+7], [16+8h, 16+8h+7]
  const int bK0  = 8 * half;         // B: K runs [8h, 8h+7], [16+8h, 16+8h+7]

  // filter-phase geometry: 16 threads per row, 8 values each
  const int frow = tid >> 4;
  const int fsub = tid & 15;

  // top-k owner state (meaningful for tid < ROWS)
  float mn = -1.0f;
  int   mnslot = 0;

  for (int c = 0; c < D_HID; c += CHUNK) {
    const int hbase = c + wave * 16;
    const unsigned short* __restrict__ whi = Whi + (size_t)(hbase + nn) * D_IN + bK0;
    const unsigned short* __restrict__ wlo = Wlo + (size_t)(hbase + nn) * D_IN + bK0;

    v8f acc_hh = {};
    v8f acc_hl = {};
    v8f acc_lh = {};
    #pragma unroll 2
    for (int k = 0; k < D_IN; k += 32) {
      Frag32 fa;
      fa.lo = *(const u32x4*)&xs_hi[nn][k + aK0];
      fa.hi = *(const u32x4*)&xs_hi[nn][k + 16 + aK0];
      v16bf a_hi = __builtin_bit_cast(v16bf, fa);
      fa.lo = *(const u32x4*)&xs_lo[nn][k + aK0];
      fa.hi = *(const u32x4*)&xs_lo[nn][k + 16 + aK0];
      v16bf a_lo = __builtin_bit_cast(v16bf, fa);

      Frag32 fb;
      fb.lo = *(const u32x4*)(whi + k);
      fb.hi = *(const u32x4*)(whi + k + 16);
      v16bf b_hi = __builtin_bit_cast(v16bf, fb);
      fb.lo = *(const u32x4*)(wlo + k);
      fb.hi = *(const u32x4*)(wlo + k + 16);
      v16bf b_lo = __builtin_bit_cast(v16bf, fb);

      acc_hh = __builtin_amdgcn_wmma_f32_16x16x32_bf16(
          false, a_hi, false, b_hi, (short)0, acc_hh, false, false);
      acc_hl = __builtin_amdgcn_wmma_f32_16x16x32_bf16(
          false, a_hi, false, b_lo, (short)0, acc_hl, false, false);
      acc_lh = __builtin_amdgcn_wmma_f32_16x16x32_bf16(
          false, a_lo, false, b_hi, (short)0, acc_lh, false, false);
    }

    const float bias = b_enc[hbase + nn];
    #pragma unroll
    for (int v = 0; v < 8; v++) {
      float z = ((acc_hh[v] + acc_hl[v]) + acc_lh[v]) + bias;
      act[mbase + v][wave * 16 + nn] = z > 0.0f ? z : 0.0f;
    }
    __syncthreads();

    // ---- phase A: parallel candidate filter (all 256 threads) ----
    {
      const float rmn = rowmin[frow];
      unsigned int m = 0;
      #pragma unroll
      for (int b = 0; b < 8; b++)
        m |= (act[frow][fsub * 8 + b] > rmn) ? (1u << b) : 0u;
      candb[frow][fsub] = (unsigned char)m;
    }
    __syncthreads();

    // ---- phase B: row owner processes only candidates ----
    if (tid < ROWS) {
      const unsigned int* words = (const unsigned int*)candb[tid];
      #pragma unroll
      for (int w = 0; w < 4; w++) {
        unsigned int m = words[w];
        while (m) {
          const int bit = __builtin_ctz(m);
          m &= m - 1;
          const int j = w * 32 + bit;
          const float z = act[tid][j];
          if (z > mn) {                       // re-check: mn rises as we insert
            topv[tid][mnslot] = z;
            topi[tid][mnslot] = c + j;
            mn = topv[tid][0]; mnslot = 0;
            #pragma unroll
            for (int s = 1; s < TOPK; s++) {
              float tv = topv[tid][s];
              if (tv < mn) { mn = tv; mnslot = s; }
            }
          }
        }
      }
      rowmin[tid] = mn;
    }
    __syncthreads();   // also protects act + rowmin for next chunk
  }

  // ---- sparse decode (coalesced: W_dec^T rows) ----
  const int drow = tid >> 4;
  const int dsub = tid & 15;
  for (int i = dsub; i < D_IN; i += 16) {
    float acc = b_dec[i];
    #pragma unroll
    for (int k = 0; k < TOPK; k++) {
      acc += topv[drow][k] * WdT[(size_t)topi[drow][k] * D_IN + i];
    }
    out[(size_t)(r0 + drow) * D_IN + i] = acc;
  }
}

// ---------------------------------------------------------------------------
// Fallback: fp32 WMMA (16x16x4) fused kernel (works with small/no workspace)
// ---------------------------------------------------------------------------
template <bool USE_TRANS>
__global__ __launch_bounds__(256) void sae_fused_f32_kernel(
    const float* __restrict__ x,
    const float* __restrict__ W_enc,
    const float* __restrict__ b_enc,
    const float* __restrict__ Wd,      // USE_TRANS ? W_dec^T : W_dec
    const float* __restrict__ b_dec,
    float* __restrict__ out)
{
  __shared__ float xs[ROWS][XS_STRIDE];
  __shared__ float act[ROWS][ACT_STRIDE];
  __shared__ float topv[ROWS][TOPK];
  __shared__ int   topi[ROWS][TOPK];

  const int tid  = (int)threadIdx.x;
  const int wave = tid >> 5;
  const int lane = tid & 31;
  const int r0   = (int)blockIdx.x * ROWS;

  {
    const float* src = x + (size_t)r0 * D_IN;
    for (int idx = tid; idx < ROWS * D_IN; idx += 256) {
      int r = idx / D_IN;
      int ccol = idx - r * D_IN;
      xs[r][ccol] = src[idx];
    }
  }
  for (int idx = tid; idx < ROWS * TOPK; idx += 256) {
    ((float*)topv)[idx] = -1.0f;
    ((int*)topi)[idx]   = 0;
  }
  __syncthreads();

  const int nn    = lane & 15;
  const int koff  = (lane >> 4) << 1;
  const int mbase = (lane >> 4) << 3;

  float mn = -1.0f;
  int   mnslot = 0;

  for (int c = 0; c < D_HID; c += CHUNK) {
    const int hbase = c + wave * 16;
    const float* __restrict__ wrow = W_enc + (size_t)(hbase + nn) * D_IN + koff;

    v8f acc0 = {};
    v8f acc1 = {};
    #pragma unroll 4
    for (int k = 0; k < D_IN; k += 8) {
      v2f a0 = *(const v2f*)&xs[nn][k + koff];
      v2f b0 = *(const v2f*)(wrow + k);
      acc0 = __builtin_amdgcn_wmma_f32_16x16x4_f32(
          false, a0, false, b0, (short)0, acc0, false, false);
      v2f a1 = *(const v2f*)&xs[nn][k + 4 + koff];
      v2f b1 = *(const v2f*)(wrow + k + 4);
      acc1 = __builtin_amdgcn_wmma_f32_16x16x4_f32(
          false, a1, false, b1, (short)0, acc1, false, false);
    }

    const float bias = b_enc[hbase + nn];
    #pragma unroll
    for (int v = 0; v < 8; v++) {
      float z = acc0[v] + acc1[v] + bias;
      act[mbase + v][wave * 16 + nn] = z > 0.0f ? z : 0.0f;
    }
    __syncthreads();

    if (tid < ROWS) {
      for (int j = 0; j < CHUNK; j++) {
        float z = act[tid][j];
        if (z > mn) {
          topv[tid][mnslot] = z;
          topi[tid][mnslot] = c + j;
          mn = topv[tid][0]; mnslot = 0;
          #pragma unroll
          for (int s = 1; s < TOPK; s++) {
            float tv = topv[tid][s];
            if (tv < mn) { mn = tv; mnslot = s; }
          }
        }
      }
    }
    __syncthreads();
  }

  const int drow = tid >> 4;
  const int dsub = tid & 15;
  for (int i = dsub; i < D_IN; i += 16) {
    float acc = b_dec[i];
    #pragma unroll
    for (int k = 0; k < TOPK; k++) {
      const float zv = topv[drow][k];
      const int   h  = topi[drow][k];
      const float w  = USE_TRANS ? Wd[(size_t)h * D_IN + i]
                                 : Wd[(size_t)i * D_HID + h];
      acc += zv * w;
    }
    out[(size_t)(r0 + drow) * D_IN + i] = acc;
  }
}

// ---------------------------------------------------------------------------
extern "C" void kernel_launch(void* const* d_in, const int* in_sizes, int n_in,
                              void* d_out, int out_size, void* d_ws, size_t ws_size,
                              hipStream_t stream) {
  const float* x     = (const float*)d_in[0];
  const float* W_enc = (const float*)d_in[1];
  const float* b_enc = (const float*)d_in[2];
  const float* W_dec = (const float*)d_in[3];
  const float* b_dec = (const float*)d_in[4];
  float* out = (float*)d_out;

  const int B = in_sizes[0] / D_IN;          // 8192
  const int nblocks = B / ROWS;              // 512

  const size_t W_ELEMS      = (size_t)D_HID * D_IN;              // 12.58M
  const size_t bf16_bytes   = W_ELEMS * sizeof(unsigned short);  // ~25 MB
  const size_t trans_bytes  = W_ELEMS * sizeof(float);           // ~50 MB
  const size_t full_bytes   = 2 * bf16_bytes + trans_bytes;      // ~100 MB

  if (ws_size >= full_bytes) {
    unsigned short* Whi = (unsigned short*)d_ws;
    unsigned short* Wlo = Whi + W_ELEMS;
    float* WdT = (float*)((char*)d_ws + 2 * bf16_bytes);
    wenc_split_kernel<<<(int)(W_ELEMS / (256 * 4)), 256, 0, stream>>>(W_enc, Whi, Wlo);
    wdec_transpose_kernel<<<dim3(D_HID / 32, D_IN / 32), dim3(32, 8), 0, stream>>>(
        W_dec, WdT);
    sae_fused_bf16_kernel<<<nblocks, 256, 0, stream>>>(
        x, Whi, Wlo, b_enc, WdT, b_dec, out);
  } else if (ws_size >= trans_bytes) {
    float* WdT = (float*)d_ws;
    wdec_transpose_kernel<<<dim3(D_HID / 32, D_IN / 32), dim3(32, 8), 0, stream>>>(
        W_dec, WdT);
    sae_fused_f32_kernel<true><<<nblocks, 256, 0, stream>>>(
        x, W_enc, b_enc, WdT, b_dec, out);
  } else {
    sae_fused_f32_kernel<false><<<nblocks, 256, 0, stream>>>(
        x, W_enc, b_enc, W_dec, b_dec, out);
  }
}
